// Bottleneck_ScaledDotProductAttention_3650722202069
// MI455X (gfx1250) — matrix operand order repaired and checked by reading, not compiled, not run on hardware
//
#include <hip/hip_runtime.h>
#include <hip/hip_bf16.h>

// ---------------- CDNA5 WMMA types ----------------
typedef __attribute__((ext_vector_type(16))) __bf16 v16bf;
typedef __attribute__((ext_vector_type(8)))  float  v8f;

union Frag16 {           // 16 bf16 = 32 bytes = 8 VGPRs (A or B operand)
    v16bf v;
    uint4 q[2];
    __bf16 h[16];
};
union Pack8 {            // 8 bf16 = 16 bytes
    uint4 q;
    __bf16 h[8];
};

__device__ inline v8f wmma_bf16(v16bf a, v16bf b, v8f c) {
    // (neg_a, A, neg_b, B, c_mod, C, reuse_a, reuse_b)
    return __builtin_amdgcn_wmma_f32_16x16x32_bf16(false, a, false, b, (short)0, c, false, false);
}

// A fragment: 16x32 bf16, row M = lane&15.
// lanes 0-15 hold K = k0+0..15 ; lanes 16-31 hold K = k0+16..31
__device__ inline v16bf load_a(const __bf16* rowK0, int grp) {
    Frag16 f;
    f.q[0] = *(const uint4*)(rowK0 + grp * 16);
    f.q[1] = *(const uint4*)(rowK0 + 8 + grp * 16);
    return f.v;
}
// B fragment: 32x16 bf16, col N = lane&15; lane holds K = k0 + 16*grp + (0..15),
// i.e. 32 contiguous bytes when the K dimension is contiguous in memory.
__device__ inline v16bf load_b(const __bf16* p) {
    Frag16 f;
    f.q[0] = *(const uint4*)(p);
    f.q[1] = *(const uint4*)(p + 8);
    return f.v;
}
__device__ inline v16bf zero_frag() {
    Frag16 f;
    f.q[0] = make_uint4(0u, 0u, 0u, 0u);
    f.q[1] = make_uint4(0u, 0u, 0u, 0u);
    return f.v;
}

#define BN_EPS 1e-5f

// ---------------- prep: fold BN scale into weights, repack [co][(kh*3+kw)*CIN+ci] ----------------
__global__ __launch_bounds__(256) void prep_w_kernel(
    const float* __restrict__ w, const float* __restrict__ g, const float* __restrict__ var,
    __bf16* __restrict__ wp, int CIN, int COUT) {
    int K9 = CIN * 9;
    long long id = (long long)blockIdx.x * 256 + threadIdx.x;
    if (id >= (long long)COUT * K9) return;
    int co = (int)(id / K9);
    int k  = (int)(id % K9);
    int pos = k / CIN, ci = k % CIN;
    int kh = pos / 3, kw = pos % 3;
    float s = g[co] * rsqrtf(var[co] + BN_EPS);
    wp[id] = (__bf16)(w[((co * CIN + ci) * 3 + kh) * 3 + kw] * s);
}

__global__ __launch_bounds__(256) void prep_bias_kernel(
    const float* __restrict__ g, const float* __restrict__ bt,
    const float* __restrict__ m, const float* __restrict__ var,
    float* __restrict__ bias, int COUT) {
    int i = blockIdx.x * 256 + threadIdx.x;
    if (i >= COUT) return;
    float s = g[i] * rsqrtf(var[i] + BN_EPS);
    bias[i] = bt[i] - m[i] * s;
}

// ---------------- x fp32 NCHW -> bf16 NHWC ----------------
__global__ __launch_bounds__(256) void cvt_x_kernel(const float* __restrict__ X,
                                                    __bf16* __restrict__ Xh) {
    long long id = (long long)blockIdx.x * 256 + threadIdx.x;
    int c = (int)(id & 255);
    long long p = id >> 8;            // b*4096 + h*64 + w
    int w  = (int)(p & 63);
    int h  = (int)((p >> 6) & 63);
    int b  = (int)(p >> 12);
    Xh[id] = (__bf16)X[(((long long)b * 256 + c) * 64 + h) * 64 + w];
}

// ---------------- implicit-GEMM conv3x3 + bias + SiLU (bf16 WMMA) ----------------
// wave = 16 pixels along w (fixed b,h) x (16*MT) output channels.
// Weight slab for each (kh,kw) is staged into LDS with CDNA5 async copies
// (GLOBAL_LOAD_ASYNC_TO_LDS_B128 / ASYNCcnt), double-buffered so the async
// engine prefetches slab s+1 while WMMAs consume slab s.
template <int CIN, int MT, bool DUAL>
__global__ __launch_bounds__(256) void conv_kernel(
    const __bf16* __restrict__ in,     // NHWC bf16 [16,64,64,CIN]
    const __bf16* __restrict__ wgt,    // [coutTotal, CIN*9] bf16, BN-folded
    const float* __restrict__ bias,
    __bf16* __restrict__ outNHWC,      // [16,64,64,coutTotal]
    __bf16* __restrict__ outNCHW,      // [16,coutTotal,64,64] (if DUAL)
    int coutTotal) {
    constexpr int H = 64, W = 64, K9 = CIN * 9;
    constexpr int SLAB  = 128 * CIN;   // bf16 elements per (kh,kw) weight slab
    constexpr int RB    = CIN / 8;     // 16B chunks per weight row
    constexpr int CHUNK = 128 * RB;    // 16B chunks per slab
    extern __shared__ __bf16 wlds[];   // 2 * SLAB (double buffer)

    int tid  = threadIdx.x;
    int wid  = tid >> 5;
    int lane = tid & 31;
    int gw   = blockIdx.x * 8 + wid;
    int wt = gw & 3;              // W/16 = 4 tiles
    int h  = (gw >> 2) & 63;
    int b  = (gw >> 8) & 15;
    int cg = gw >> 12;            // grids sized so cg < coutTotal/(16*MT)
    int co0 = cg * (16 * MT);
    int col = lane & 15, grp = lane >> 4;
    int w0 = wt * 16;

    // async-stage weight slab s (rows co0..co0+127, K-range s*CIN..s*CIN+CIN)
    auto stage = [&](int s) {
        const __bf16* wsrc = wgt + (size_t)co0 * K9 + s * CIN;
        __bf16* dst = wlds + (s & 1) * SLAB;
#pragma unroll
        for (int i = tid; i < CHUNK; i += 256) {
            int row = i / RB;
            int c8  = (i % RB) * 8;
            unsigned loff = (unsigned)(uintptr_t)(dst + row * CIN + c8);
            const __bf16* gp = wsrc + (size_t)row * K9 + c8;
            asm volatile("global_load_async_to_lds_b128 %0, %1, off"
                         :: "v"(loff), "v"(gp) : "memory");
        }
    };

    v8f acc[MT];
#pragma unroll
    for (int i = 0; i < MT; i++) { v8f z = {}; acc[i] = z; }

    stage(0);
    for (int s = 0; s < 9; s++) {
        asm volatile("s_wait_asynccnt 0x0" ::: "memory");  // slab s landed (per-wave)
        __syncthreads();                                   // visible to all waves
        if (s < 8) stage(s + 1);                           // prefetch next slab

        int kh = s / 3, kw = s % 3;
        int ih = h + kh - 1;
        if (ih >= 0 && ih < H) {
            const __bf16* rowp = in + (size_t)((b * H + ih) * W) * CIN;
            int iw = w0 + col + kw - 1;
            bool valid = (iw >= 0) && (iw < W);
            const __bf16* pix = rowp + (size_t)(valid ? iw : 0) * CIN + grp * 16;
            const __bf16* wbase = wlds + (s & 1) * SLAB;
#pragma unroll 4
            for (int ci0 = 0; ci0 < CIN; ci0 += 32) {
                v16bf bf = valid ? load_b(pix + ci0) : zero_frag();
#pragma unroll
                for (int mt = 0; mt < MT; mt++) {
                    const __bf16* lp = wbase + (16 * mt + col) * CIN + ci0;
                    Frag16 af;                        // ds_load_b128 x2
                    af.q[0] = *(const uint4*)(lp + grp * 16);
                    af.q[1] = *(const uint4*)(lp + 8 + grp * 16);
                    acc[mt] = wmma_bf16(af.v, bf, acc[mt]);
                }
            }
        }
        // no trailing barrier needed: a wave only reaches the next iteration's
        // barrier after finishing this compute, and stage(s+1) targets the
        // other buffer half.
    }

    // epilogue: bias + SiLU, store NHWC (and optionally NCHW)
    long long n_sp = (long long)(b * H + h) * W + w0 + col;   // pixel index
#pragma unroll
    for (int mt = 0; mt < MT; mt++) {
        int coB = co0 + 16 * mt + 8 * grp;                    // D: M = r + 8*grp
        Pack8 p;
#pragma unroll
        for (int r = 0; r < 8; r++) {
            float y = acc[mt][r] + bias[coB + r];
            float s = y / (1.0f + __expf(-y));                // SiLU
            p.h[r] = (__bf16)s;
            if (DUAL)
                outNCHW[((long long)b * coutTotal + coB + r) * (H * W) + h * W + w0 + col] =
                    (__bf16)s;
        }
        *(uint4*)(outNHWC + n_sp * coutTotal + coB) = p.q;
    }
}

// ---------------- scores = (Y @ Y^T) / 16 ; Y = [C=256, N=4096] bf16 NCHW per batch ----------------
__global__ __launch_bounds__(128) void scores_kernel(const __bf16* __restrict__ Y,
                                                     float* __restrict__ S) {
    constexpr int C = 256, N = 4096;
    int wid = threadIdx.x >> 5, lane = threadIdx.x & 31;
    int col = lane & 15, grp = lane >> 4;
    int b  = blockIdx.z;
    int c0 = blockIdx.x * 64;
    int d0 = blockIdx.y * 64 + wid * 16;
    const __bf16* Yb = Y + (long long)b * C * N;

    v8f acc[4];
#pragma unroll
    for (int i = 0; i < 4; i++) { v8f z = {}; acc[i] = z; }

    const __bf16* bp = Yb + (long long)(d0 + col) * N + grp * 16;
#pragma unroll 2
    for (int n0 = 0; n0 < N; n0 += 32) {
        v16bf bf = load_b(bp + n0);
#pragma unroll
        for (int mt = 0; mt < 4; mt++) {
            const __bf16* ap = Yb + (long long)(c0 + 16 * mt + col) * N + n0;
            acc[mt] = wmma_bf16(load_a(ap, grp), bf, acc[mt]);
        }
    }
    const float invT = 1.0f / 16.0f;   // 1/sqrt(256)
#pragma unroll
    for (int mt = 0; mt < 4; mt++) {
        int cB = c0 + 16 * mt + 8 * grp;
#pragma unroll
        for (int r = 0; r < 8; r++)
            S[((long long)b * C + cB + r) * C + d0 + col] = acc[mt][r] * invT;
    }
}

// ---------------- softmax over last dim (256), one wave per row, bf16 out ----------------
__global__ __launch_bounds__(256) void softmax_kernel(const float* __restrict__ S,
                                                      __bf16* __restrict__ A) {
    int wid = threadIdx.x >> 5, lane = threadIdx.x & 31;
    int row = blockIdx.x * 8 + wid;          // 16*256 rows
    const float* sp = S + (long long)row * 256;
    float v[8];
    float mx = -1e30f;
#pragma unroll
    for (int j = 0; j < 8; j++) { v[j] = sp[lane + 32 * j]; mx = fmaxf(mx, v[j]); }
#pragma unroll
    for (int o = 16; o > 0; o >>= 1) mx = fmaxf(mx, __shfl_xor(mx, o, 32));
    float sum = 0.0f;
#pragma unroll
    for (int j = 0; j < 8; j++) { v[j] = __expf(v[j] - mx); sum += v[j]; }
#pragma unroll
    for (int o = 16; o > 0; o >>= 1) sum += __shfl_xor(sum, o, 32);
    float inv = 1.0f / sum;
    __bf16* ap = A + (long long)row * 256;
#pragma unroll
    for (int j = 0; j < 8; j++) ap[lane + 32 * j] = (__bf16)(v[j] * inv);
}

// ---------------- out = x + attn @ Y ; attn [C,C] bf16, Yt = NHWC bf16 [n, d] ----------------
__global__ __launch_bounds__(256) void attnout_kernel(const __bf16* __restrict__ A,
                                                      const __bf16* __restrict__ Yt,
                                                      const float* __restrict__ X,
                                                      float* __restrict__ Out) {
    constexpr int C = 256, N = 4096;
    int wid = threadIdx.x >> 5, lane = threadIdx.x & 31;
    int col = lane & 15, grp = lane >> 4;
    int b  = blockIdx.z;
    int n0 = (blockIdx.x * 8 + wid) * 16;
    int c0 = blockIdx.y * 128;

    v8f acc[8];
#pragma unroll
    for (int i = 0; i < 8; i++) { v8f z = {}; acc[i] = z; }

    const __bf16* bp = Yt + ((long long)b * N + n0 + col) * C + grp * 16;
    const __bf16* Ab = A + (long long)b * C * C;
#pragma unroll
    for (int d0 = 0; d0 < C; d0 += 32) {
        v16bf bf = load_b(bp + d0);
#pragma unroll
        for (int mt = 0; mt < 8; mt++) {
            const __bf16* ap = Ab + (long long)(c0 + 16 * mt + col) * C + d0;
            acc[mt] = wmma_bf16(load_a(ap, grp), bf, acc[mt]);
        }
    }
#pragma unroll
    for (int mt = 0; mt < 8; mt++) {
        int cB = c0 + 16 * mt + 8 * grp;
#pragma unroll
        for (int r = 0; r < 8; r++) {
            long long idx = ((long long)b * C + cB + r) * N + n0 + col;
            Out[idx] = X[idx] + acc[mt][r];
        }
    }
}

// ---------------- host launch ----------------
extern "C" void kernel_launch(void* const* d_in, const int* in_sizes, int n_in,
                              void* d_out, int out_size, void* d_ws, size_t ws_size,
                              hipStream_t stream) {
    (void)in_sizes; (void)n_in; (void)out_size; (void)ws_size;
    const float* x  = (const float*)d_in[0];
    const float* w1 = (const float*)d_in[1];
    const float* g1 = (const float*)d_in[2];
    const float* b1 = (const float*)d_in[3];
    const float* m1 = (const float*)d_in[4];
    const float* v1 = (const float*)d_in[5];
    const float* w2 = (const float*)d_in[6];
    const float* g2 = (const float*)d_in[7];
    const float* b2 = (const float*)d_in[8];
    const float* m2 = (const float*)d_in[9];
    const float* v2 = (const float*)d_in[10];
    float* out = (float*)d_out;

    // workspace carve (256B aligned)
    char* ws = (char*)d_ws;
    auto carve = [&](size_t bytes) {
        char* p = ws;
        ws += (bytes + 255) & ~(size_t)255;
        return p;
    };
    __bf16* w1p   = (__bf16*)carve(128 * 2304 * sizeof(__bf16));
    float*  bias1 = (float*) carve(128 * sizeof(float));
    __bf16* w2p   = (__bf16*)carve(256 * 1152 * sizeof(__bf16));
    float*  bias2 = (float*) carve(256 * sizeof(float));
    __bf16* xh    = (__bf16*)carve((size_t)16 * 4096 * 256 * sizeof(__bf16)); // NHWC
    __bf16* y1h   = (__bf16*)carve((size_t)16 * 4096 * 128 * sizeof(__bf16)); // NHWC
    __bf16* y2n   = (__bf16*)carve((size_t)16 * 256 * 4096 * sizeof(__bf16)); // NCHW
    __bf16* y2t   = (__bf16*)carve((size_t)16 * 4096 * 256 * sizeof(__bf16)); // NHWC
    float*  sc    = (float*) carve((size_t)16 * 256 * 256 * sizeof(float));
    __bf16* attn  = (__bf16*)carve((size_t)16 * 256 * 256 * sizeof(__bf16));

    // 1) weight/bias prep (BN folded)
    prep_w_kernel<<<(128 * 2304 + 255) / 256, 256, 0, stream>>>(w1, g1, v1, w1p, 256, 128);
    prep_bias_kernel<<<1, 256, 0, stream>>>(g1, b1, m1, v1, bias1, 128);
    prep_w_kernel<<<(256 * 1152 + 255) / 256, 256, 0, stream>>>(w2, g2, v2, w2p, 128, 256);
    prep_bias_kernel<<<1, 256, 0, stream>>>(g2, b2, m2, v2, bias2, 256);

    // 2) x -> NHWC bf16
    cvt_x_kernel<<<(16 * 4096 * 256) / 256, 256, 0, stream>>>(x, xh);

    // 3) conv1: 256->128. waves = 16*64*4 = 4096 -> 512 blocks; dyn-LDS 2*64KB
    conv_kernel<256, 8, false><<<512, 256, 2 * 128 * 256 * sizeof(__bf16), stream>>>(
        xh, w1p, bias1, y1h, nullptr, 128);

    // 4) conv2: 128->256, dual NHWC+NCHW out. waves = 8192 -> 1024 blocks; dyn-LDS 2*32KB
    conv_kernel<128, 8, true><<<1024, 256, 2 * 128 * 128 * sizeof(__bf16), stream>>>(
        y1h, w2p, bias2, y2t, y2n, 256);

    // 5) scores = Y Y^T / 16   (64x64 tile per block, 4 waves)
    scores_kernel<<<dim3(4, 4, 16), 128, 0, stream>>>(y2n, sc);

    // 6) softmax rows (16*256 rows, 8 waves/block)
    softmax_kernel<<<512, 256, 0, stream>>>(sc, attn);

    // 7) out = x + attn @ Y
    attnout_kernel<<<dim3(32, 2, 16), 256, 0, stream>>>(attn, y2t, x, out);
}
